// Model_53300544143858
// MI455X (gfx1250) — compile-verified, matrix operand order checked by reading, not run on hardware
//
#include <hip/hip_runtime.h>
#include <hip/hip_bf16.h>

// Graph-WaveNet forward for MI455X (gfx1250, wave32, WMMA).
// Channel-fastest activation layout [col][C] so every WMMA A/B fragment is two
// contiguous 16B loads and every epilogue is a vector store. All contractions
// run on v_wmma_f32_16x16x32_f16 (f16 operands, f32 accumulate).

typedef __attribute__((ext_vector_type(16))) _Float16 v16h;
typedef __attribute__((ext_vector_type(8)))  _Float16 v8h;
typedef __attribute__((ext_vector_type(8)))  float    v8f;

#define NB    4
#define NN    207
#define NSEQ  96
#define NPRED 12
#define TT0   373
#define PADL  277
#define BNROW 828L            // NB*NN
#define COLS0 308844L         // BNROW*TT0
#define TS    361             // final skip time length
#define SCOLS 298908L         // BNROW*TS
#define OUTNT 74727L          // NN*TS
#define ADLD  224             // padded v-stride of transposed adjacency
#define EPSBN 1e-5f

__device__ inline v8f zero8() {
  v8f z;
#pragma unroll
  for (int i = 0; i < 8; ++i) z[i] = 0.f;
  return z;
}
__device__ inline v16h zero16h() {
  v16h z;
#pragma unroll
  for (int i = 0; i < 16; ++i) z[i] = (_Float16)0.f;
  return z;
}
__device__ inline v16h frag_cat(v8h lo, v8h hi) {
  v16h f;
#pragma unroll
  for (int i = 0; i < 8; ++i) { f[i] = lo[i]; f[8 + i] = hi[i]; }
  return f;
}

// A fragment (16x32 f16) from row-major memory: elem (m,k) at base[m*ld+k].
// Lane m=l&15; low lanes hold K {0..7,16..23}, high lanes K {8..15,24..31}.
// Each lane = two contiguous 16B loads. ld must be a multiple of 16 halves.
__device__ inline v16h fragA(const _Float16* base, int ld, int lane) {
  int m = lane & 15;
  int koff = (lane >= 16) ? 8 : 0;
  const v8h* p0 = (const v8h*)(base + m * ld + koff);
  const v8h* p1 = (const v8h*)(base + m * ld + 16 + koff);
  return frag_cat(*p0, *p1);
}

// B fragment from channel-fastest activation [col][ld]: elem (k,n) at act[(col0+n)*ld + k0 + k].
__device__ inline v16h fragB_cf(const _Float16* act, long col0, long cols, int ld, int k0,
                                int lane) {
  long n = col0 + (lane & 15);
  int koff = (lane >= 16) ? 8 : 0;
  if (n < cols) {
    const v8h* p0 = (const v8h*)(act + n * (long)ld + k0 + koff);
    const v8h* p1 = (const v8h*)(act + n * (long)ld + k0 + 16 + koff);
    return frag_cat(*p0, *p1);
  }
  return zero16h();
}

#define WMMA_F16(a, b, c) \
  __builtin_amdgcn_wmma_f32_16x16x32_f16(false, (a), false, (b), (short)0, (c), false, false)

// ---------------- small packing kernels ----------------

__global__ void k_cvt_f16(const float* __restrict__ src, _Float16* __restrict__ dst, int n) {
  int i = blockIdx.x * 256 + threadIdx.x;
  if (i < n) dst[i] = (_Float16)src[i];
}

__global__ void k_zero_h(_Float16* __restrict__ p, int n) {
  int i = blockIdx.x * 256 + threadIdx.x;
  if (i < n) p[i] = (_Float16)0.f;
}

// wcat[i][o][k]: o<32 -> Wf rows, o>=32 -> Wg rows; k = tap*32 + c
__global__ void k_pack_wcat(const float* __restrict__ Wf, const float* __restrict__ Wg,
                            _Float16* __restrict__ wcat) {
  int idx = blockIdx.x * 256 + threadIdx.x;
  if (idx >= 8 * 64 * 64) return;
  int k = idx & 63, o = (idx >> 6) & 63, i = idx >> 12;
  int tap = k >> 5, c = k & 31;
  float v = (o < 32) ? Wf[(((i * 32 + o) * 32 + c) * 2) + tap]
                     : Wg[(((i * 32 + (o - 32)) * 32 + c) * 2) + tap];
  wcat[idx] = (_Float16)v;
}

// W_end2 (12x512) zero-padded to 16x512
__global__ void k_pack_w2(const float* __restrict__ W2, _Float16* __restrict__ w2h) {
  int idx = blockIdx.x * 256 + threadIdx.x;
  if (idx >= 16 * 512) return;
  int k = idx & 511, p = idx >> 9;
  w2h[idx] = (_Float16)((p < 12) ? W2[p * 512 + k] : 0.f);
}

// adp = softmax(relu(E1@E2), axis=1); stored TRANSPOSED + padded: adpT[w*ADLD + v]
__global__ void k_adp(const float* __restrict__ E1, const float* __restrict__ E2,
                      _Float16* __restrict__ adpT) {
  __shared__ float red[256];
  int v = blockIdx.x, w = threadIdx.x;
  float val = 0.f;
  if (w < NN) {
    float s = 0.f;
    for (int r = 0; r < 10; ++r) s += E1[v * 10 + r] * E2[r * NN + w];
    val = fmaxf(s, 0.f);
  }
  red[w] = (w < NN) ? val : -3.4e38f;
  __syncthreads();
  for (int s = 128; s > 0; s >>= 1) {
    if (w < s) red[w] = fmaxf(red[w], red[w + s]);
    __syncthreads();
  }
  float mx = red[0];
  __syncthreads();
  float e = (w < NN) ? __expf(val - mx) : 0.f;
  red[w] = e;
  __syncthreads();
  for (int s = 128; s > 0; s >>= 1) {
    if (w < s) red[w] += red[w + s];
    __syncthreads();
  }
  if (w < NN) adpT[w * ADLD + v] = (_Float16)(e / red[0]);
}

// ---------------- start conv (1 -> 32, left zero-pad to T=373), layout [col][32] ----------------

__global__ void k_start(const float* __restrict__ in, const float* __restrict__ sW,
                        const float* __restrict__ sb, float* __restrict__ xcur,
                        _Float16* __restrict__ xh) {
  long idx = (long)blockIdx.x * 256 + threadIdx.x;
  if (idx >= 32 * COLS0) return;
  long col = idx >> 5;
  int c = (int)(idx & 31);
  long bn = col / TT0;
  int t = (int)(col % TT0);
  int b = (int)(bn / NN), n = (int)(bn % NN);
  int tin = t - PADL;
  float v = sb[c];
  if (tin >= 0) v += sW[c] * in[((long)b * NSEQ + tin) * NN + n];
  xcur[idx] = v;
  xh[idx] = (_Float16)v;
}

// ---------------- dilated conv + gate: x = tanh(f)*sigmoid(g) -> H channels 0..31 ----------------

__global__ void k_dilconv(const _Float16* __restrict__ wcat, const float* __restrict__ bfp,
                          const float* __restrict__ bgp, const _Float16* __restrict__ xh,
                          int T, int d, _Float16* __restrict__ H, long cols) {
  int lane = threadIdx.x & 31;
  int Tp = T - d;
  long col0 = (long)blockIdx.x * 16;
  int y = blockIdx.y;  // filt rows y*16.., gate rows 32+y*16..
  long j = col0 + (lane & 15);
  bool colok = (j < cols);
  long base_in = 0;
  if (colok) {
    long bn = j / Tp;
    int t = (int)(j % Tp);
    base_in = bn * T + t;
  }
  int koff = (lane >= 16) ? 8 : 0;
  v8f accF = zero8(), accG = zero8();
#pragma unroll
  for (int k0 = 0; k0 < 64; k0 += 32) {
    v16h bfrag = zero16h();
    if (colok) {
      long bin = base_in + (k0 ? d : 0);  // tap 1 shifts time by d
      const v8h* p0 = (const v8h*)(xh + bin * 32 + koff);
      const v8h* p1 = (const v8h*)(xh + bin * 32 + 16 + koff);
      bfrag = frag_cat(*p0, *p1);
    }
    v16h af = fragA(wcat + (y * 16) * 64 + k0, 64, lane);
    v16h ag = fragA(wcat + (32 + y * 16) * 64 + k0, 64, lane);
    accF = WMMA_F16(af, bfrag, accF);
    accG = WMMA_F16(ag, bfrag, accG);
  }
  if (colok) {
    int mbase = (lane >= 16) ? 8 : 0;
    v8h hv;
#pragma unroll
    for (int r = 0; r < 8; ++r) {
      int o = y * 16 + mbase + r;
      float f = tanhf(accF[r] + bfp[o]);
      float g = accG[r] + bgp[o];
      g = 1.f / (1.f + __expf(-g));
      hv[r] = (_Float16)(f * g);
    }
    *(v8h*)(H + j * 96 + y * 16 + mbase) = hv;
  }
}

// ---------------- skip projection 32->256, accumulated with time alignment ----------------

__global__ void k_skip(const _Float16* __restrict__ ws, const float* __restrict__ bs,
                       const _Float16* __restrict__ H, long cols, int Tp,
                       float* __restrict__ skip) {
  int lane = threadIdx.x & 31;
  long col0 = (long)blockIdx.x * 16;
  int row0 = blockIdx.y * 16;
  v8f acc = zero8();
  v16h a = fragA(ws + row0 * 32, 32, lane);
  v16h b = fragB_cf(H, col0, cols, 96, 0, lane);
  acc = WMMA_F16(a, b, acc);
  long j = col0 + (lane & 15);
  if (j < cols) {
    long bn = j / Tp;
    int t = (int)(j % Tp);
    int toff = t - (Tp - TS);
    if (toff >= 0) {
      int mbase = (lane >= 16) ? 8 : 0;
      float* dst = skip + (bn * TS + toff) * 256 + row0 + mbase;
      v8f old = *(const v8f*)dst;
#pragma unroll
      for (int r = 0; r < 8; ++r) old[r] += acc[r] + bs[row0 + mbase + r];
      *(v8f*)dst = old;
    }
  }
}

// ---------------- graph diffusion: H[dst] = H[src] x adp  (per (b,t) 32x207 @ 207x207) ----------

__global__ void k_diffuse(_Float16* __restrict__ H, long cols, int Tp,
                          const _Float16* __restrict__ adpT, int srcRow, int dstRow) {
  int lane = threadIdx.x & 31;
  int w0 = blockIdx.x * 16;
  int c0 = blockIdx.y * 16;
  long bt = blockIdx.z;
  int b = (int)(bt / Tp), t = (int)(bt % Tp);
  int koff = (lane >= 16) ? 8 : 0;
  int m = lane & 15;
  v8f acc = zero8();
  for (int v0 = 0; v0 < NN; v0 += 32) {
    // A: (m=c, k=v) gathered (stride Tp*96 between v) with v-guards
    v16h a = zero16h();
#pragma unroll
    for (int jj = 0; jj < 8; ++jj) {
      int k = ((jj & 3) << 1) + koff + ((jj & 4) << 2);
      int v = v0 + k;
      if (v < NN)
        a[2 * jj] = H[(((long)b * NN + v) * Tp + t) * 96 + srcRow + c0 + m];
      if (v + 1 < NN)
        a[2 * jj + 1] = H[(((long)b * NN + v + 1) * Tp + t) * 96 + srcRow + c0 + m];
    }
    // B: (k=v, n=w) from transposed zero-padded adjacency: two contiguous 16B loads
    const v8h* q0 = (const v8h*)(adpT + (w0 + m) * ADLD + v0 + koff);
    const v8h* q1 = (const v8h*)(adpT + (w0 + m) * ADLD + v0 + 16 + koff);
    v16h bb = frag_cat(*q0, *q1);
    acc = WMMA_F16(a, bb, acc);
  }
  if (w0 + m < NN) {
    int mbase = (lane >= 16) ? 8 : 0;
    v8h hv;
#pragma unroll
    for (int r = 0; r < 8; ++r) hv[r] = (_Float16)acc[r];
    *(v8h*)(H + (((long)b * NN + (w0 + m)) * Tp + t) * 96 + dstRow + c0 + mbase) = hv;
  }
}

// ---------------- GCN mix 96->32 + residual -> xbn (pre-BN), layout [col][32] ----------------

__global__ void k_gcn(const _Float16* __restrict__ wgc, const float* __restrict__ bgc,
                      const _Float16* __restrict__ H, long cols, int Tp, int d,
                      const float* __restrict__ xres, float* __restrict__ xbn) {
  int lane = threadIdx.x & 31;
  long col0 = (long)blockIdx.x * 16;
  int row0 = blockIdx.y * 16;
  v8f acc = zero8();
#pragma unroll
  for (int k0 = 0; k0 < 96; k0 += 32) {
    v16h a = fragA(wgc + row0 * 96 + k0, 96, lane);
    v16h b = fragB_cf(H, col0, cols, 96, k0, lane);
    acc = WMMA_F16(a, b, acc);
  }
  long j = col0 + (lane & 15);
  if (j < cols) {
    long bn = j / Tp;
    int t = (int)(j % Tp);
    long rescol = bn * (Tp + d) + t + d;  // residual[..., -T':]
    int mbase = (lane >= 16) ? 8 : 0;
    v8f res = *(const v8f*)(xres + rescol * 32 + row0 + mbase);
    v8f outv;
#pragma unroll
    for (int r = 0; r < 8; ++r) outv[r] = acc[r] + bgc[row0 + mbase + r] + res[r];
    *(v8f*)(xbn + j * 32 + row0 + mbase) = outv;
  }
}

// ---------------- BatchNorm (training-mode stats), two-stage deterministic reduce ----------------

__global__ void k_bn_part(const float* __restrict__ xbn, long cols, float* __restrict__ part) {
  __shared__ float s1[256], s2[256];
  int tid = threadIdx.x;
  int c = tid & 31, g = tid >> 5;  // lanes=channels -> coalesced 128B rows
  float a = 0.f, b = 0.f;
  for (long col = blockIdx.x * 8 + g; col < cols; col += 512) {
    float v = xbn[col * 32 + c];
    a += v;
    b += v * v;
  }
  s1[tid] = a;
  s2[tid] = b;
  __syncthreads();
  for (int s = 128; s >= 32; s >>= 1) {
    if (tid < s) { s1[tid] += s1[tid + s]; s2[tid] += s2[tid + s]; }
    __syncthreads();
  }
  if (tid < 32) {
    part[blockIdx.x * 64 + tid] = s1[tid];
    part[blockIdx.x * 64 + 32 + tid] = s2[tid];
  }
}

__global__ void k_bn_final(const float* __restrict__ part, long cols, float* __restrict__ stats) {
  __shared__ float s[64];
  int tid = threadIdx.x;  // 64 threads: [0,32)=sum, [32,64)=sumsq
  float a = 0.f;
  for (int blk = 0; blk < 64; ++blk) a += part[blk * 64 + tid];
  s[tid] = a;
  __syncthreads();
  if (tid < 32) {
    float m = s[tid] / (float)cols;
    stats[tid] = m;
    stats[32 + tid] = s[32 + tid] / (float)cols - m * m;
  }
}

__global__ void k_bn_apply(const float* __restrict__ xbn, long cols,
                           const float* __restrict__ stats, const float* __restrict__ gam,
                           const float* __restrict__ bet, float* __restrict__ xcur,
                           _Float16* __restrict__ xh) {
  long idx = (long)blockIdx.x * 256 + threadIdx.x;
  if (idx >= 32 * cols) return;
  int c = (int)(idx & 31);
  float m = stats[c], v = stats[32 + c];
  float y = (xbn[idx] - m) * rsqrtf(v + EPSBN) * gam[c] + bet[c];
  xcur[idx] = y;
  xh[idx] = (_Float16)y;
}

// ---------------- head: relu(skip) -> 512 (relu) -> 12, fused per 16-col tile ----------------

// B fragment directly from f32 skip [scol][256] with fused ReLU + f16 convert
__device__ inline v16h fragB_skip(const float* __restrict__ skip, long col0, int k0, int lane) {
  long n = col0 + (lane & 15);
  int koff = (lane >= 16) ? 8 : 0;
  v16h f = zero16h();
  if (n < SCOLS) {
    v8f r0 = *(const v8f*)(skip + n * 256 + k0 + koff);
    v8f r1 = *(const v8f*)(skip + n * 256 + k0 + 16 + koff);
#pragma unroll
    for (int i = 0; i < 8; ++i) {
      f[i] = (_Float16)fmaxf(r0[i], 0.f);
      f[8 + i] = (_Float16)fmaxf(r1[i], 0.f);
    }
  }
  return f;
}

__global__ void k_head(const _Float16* __restrict__ w1h, const float* __restrict__ b1,
                       const _Float16* __restrict__ w2h, const float* __restrict__ b2,
                       const float* __restrict__ skip, float* __restrict__ out) {
  __shared__ _Float16 Y1t[16 * 512];  // Bt[n][h] of relu(W1*relu(skip)+b1)
  int lane = threadIdx.x;
  long col0 = (long)blockIdx.x * 16;
  int nloc = lane & 15;
  int mbase = (lane >= 16) ? 8 : 0;
  // hoist the 8 B fragments (K=256) once for all 32 row tiles of GEMM1
  v16h barr[8];
#pragma unroll
  for (int kk = 0; kk < 8; ++kk) barr[kk] = fragB_skip(skip, col0, kk * 32, lane);
  for (int rt = 0; rt < 32; ++rt) {
    v8f acc = zero8();
#pragma unroll
    for (int kk = 0; kk < 8; ++kk) {
      v16h a = fragA(w1h + (rt * 16) * 256 + kk * 32, 256, lane);
      acc = WMMA_F16(a, barr[kk], acc);
    }
    v8h yv;
#pragma unroll
    for (int r = 0; r < 8; ++r)
      yv[r] = (_Float16)fmaxf(acc[r] + b1[rt * 16 + mbase + r], 0.f);
    *(v8h*)(Y1t + nloc * 512 + rt * 16 + mbase) = yv;
  }
  __syncthreads();
  v8f acc2 = zero8();
#pragma unroll
  for (int k0 = 0; k0 < 512; k0 += 32) {
    v16h a = fragA(w2h + k0, 512, lane);
    v16h b = fragA(Y1t + k0, 512, lane);  // Bt[n][h]: same contiguous-run loader
    acc2 = WMMA_F16(a, b, acc2);
  }
  long j = col0 + nloc;
  if (j < SCOLS) {
    long bb = j / OUTNT;
    long rem = j % OUTNT;
#pragma unroll
    for (int r = 0; r < 8; ++r) {
      int p = mbase + r;
      if (p < NPRED) out[(bb * NPRED + p) * OUTNT + rem] = acc2[r] + b2[p];
    }
  }
}

// ---------------- host ----------------

extern "C" void kernel_launch(void* const* d_in, const int* in_sizes, int n_in,
                              void* d_out, int out_size, void* d_ws, size_t ws_size,
                              hipStream_t stream) {
  (void)in_sizes; (void)n_in; (void)out_size; (void)ws_size;
  const float* input  = (const float*)d_in[0];
  const float* startW = (const float*)d_in[5];
  const float* startB = (const float*)d_in[6];
  const float* Wf     = (const float*)d_in[7];
  const float* bfw    = (const float*)d_in[8];
  const float* Wg     = (const float*)d_in[9];
  const float* bgw    = (const float*)d_in[10];
  const float* Ws     = (const float*)d_in[11];
  const float* bsw    = (const float*)d_in[12];
  const float* Wgc    = (const float*)d_in[13];
  const float* bgcw   = (const float*)d_in[14];
  const float* gam    = (const float*)d_in[15];
  const float* bet    = (const float*)d_in[16];
  const float* nv1    = (const float*)d_in[17];
  const float* nv2    = (const float*)d_in[18];
  const float* W1     = (const float*)d_in[19];
  const float* b1     = (const float*)d_in[20];
  const float* W2     = (const float*)d_in[21];
  const float* b2     = (const float*)d_in[22];
  float* out = (float*)d_out;

  char* wsp = (char*)d_ws;
  size_t off = 0;
  auto alloc = [&](size_t bytes) -> void* {
    void* p = wsp + off;
    off = (off + bytes + 255) & ~(size_t)255;
    return p;
  };
  float*    xcur  = (float*)alloc(32 * COLS0 * 4);      // [col][32]
  _Float16* xh    = (_Float16*)alloc(32 * COLS0 * 2);   // [col][32]
  float*    xbn   = (float*)alloc(32 * COLS0 * 4);      // [col][32]
  _Float16* H     = (_Float16*)alloc(96 * COLS0 * 2);   // [col][96]
  float*    skip  = (float*)alloc(256 * SCOLS * 4);     // [scol][256]
  _Float16* adpT  = (_Float16*)alloc((size_t)208 * ADLD * 2);
  _Float16* wcat  = (_Float16*)alloc(8 * 64 * 64 * 2);
  _Float16* wsh   = (_Float16*)alloc(8 * 256 * 32 * 2);
  _Float16* wgch  = (_Float16*)alloc(8 * 32 * 96 * 2);
  _Float16* w1h   = (_Float16*)alloc(512 * 256 * 2);
  _Float16* w2h   = (_Float16*)alloc(16 * 512 * 2);
  float*    part  = (float*)alloc(64 * 64 * 4);
  float*    stats = (float*)alloc(64 * 4);

  // weight packing / f16 conversion
  k_pack_wcat<<<(8 * 64 * 64 + 255) / 256, 256, 0, stream>>>(Wf, Wg, wcat);
  k_cvt_f16<<<(8 * 256 * 32 + 255) / 256, 256, 0, stream>>>(Ws, wsh, 8 * 256 * 32);
  k_cvt_f16<<<(8 * 32 * 96 + 255) / 256, 256, 0, stream>>>(Wgc, wgch, 8 * 32 * 96);
  k_cvt_f16<<<(512 * 256 + 255) / 256, 256, 0, stream>>>(W1, w1h, 512 * 256);
  k_pack_w2<<<(16 * 512 + 255) / 256, 256, 0, stream>>>(W2, w2h);
  k_zero_h<<<(208 * ADLD + 255) / 256, 256, 0, stream>>>(adpT, 208 * ADLD);
  k_adp<<<NN, 256, 0, stream>>>(nv1, nv2, adpT);

  // zero skip accumulator, embed input
  hipMemsetAsync(skip, 0, (size_t)256 * SCOLS * 4, stream);
  k_start<<<(unsigned)((32 * COLS0 + 255) / 256), 256, 0, stream>>>(input, startW, startB, xcur,
                                                                    xh);

  static const int dils[8] = {1, 2, 1, 2, 1, 2, 1, 2};
  int T = TT0;
  for (int i = 0; i < 8; ++i) {
    int d = dils[i];
    int Tp = T - d;
    long cols = BNROW * (long)Tp;
    unsigned colTiles = (unsigned)((cols + 15) / 16);
    k_dilconv<<<dim3(colTiles, 2), 32, 0, stream>>>(wcat + (size_t)i * 64 * 64, bfw + i * 32,
                                                    bgw + i * 32, xh, T, d, H, cols);
    k_skip<<<dim3(colTiles, 16), 32, 0, stream>>>(wsh + (size_t)i * 256 * 32, bsw + i * 256, H,
                                                  cols, Tp, skip);
    k_diffuse<<<dim3(13, 2, (unsigned)(NB * Tp)), 32, 0, stream>>>(H, cols, Tp, adpT, 0, 32);
    k_diffuse<<<dim3(13, 2, (unsigned)(NB * Tp)), 32, 0, stream>>>(H, cols, Tp, adpT, 32, 64);
    k_gcn<<<dim3(colTiles, 2), 32, 0, stream>>>(wgch + (size_t)i * 32 * 96, bgcw + i * 32, H, cols,
                                                Tp, d, xcur, xbn);
    k_bn_part<<<64, 256, 0, stream>>>(xbn, cols, part);
    k_bn_final<<<1, 64, 0, stream>>>(part, cols, stats);
    k_bn_apply<<<(unsigned)((32 * cols + 255) / 256), 256, 0, stream>>>(
        xbn, cols, stats, gam + i * 32, bet + i * 32, xcur, xh);
    T = Tp;
  }

  k_head<<<(unsigned)((SCOLS + 15) / 16), 32, 0, stream>>>(w1h, b1, w2h, b2, skip, out);
}
